// ContextQueryAttention_44066364457466
// MI455X (gfx1250) — matrix operand order repaired and checked by reading, not compiled, not run on hardware
//
#include <hip/hip_runtime.h>
#include <cstdint>

typedef unsigned short u16;
typedef long long ll;

typedef __attribute__((ext_vector_type(16))) __bf16 v16bf;
typedef __attribute__((ext_vector_type(8)))  float  v8f;

struct alignas(16) B16 { uint32_t w[4]; };     // POD 16-byte chunk (unions OK)
union Frag16 { v16bf v; B16 b[2]; };

// ---- CDNA5 async global->LDS copy (ASYNCcnt path), with safe fallback ----
#if defined(__has_builtin)
#  if __has_builtin(__builtin_amdgcn_global_load_async_to_lds_b128)
#    define USE_ASYNC_LDS 1
#  endif
#endif
#ifndef USE_ASYNC_LDS
#  define USE_ASYNC_LDS 0
#endif

#if USE_ASYNC_LDS
#  if __has_builtin(__builtin_amdgcn_s_wait_asynccnt)
#    define WAIT_ASYNC() __builtin_amdgcn_s_wait_asynccnt(0)
#  else
#    define WAIT_ASYNC() asm volatile("s_wait_asynccnt 0x0" ::: "memory")
#  endif
typedef __attribute__((ext_vector_type(4))) int v4i;
typedef __attribute__((address_space(1))) void as1_void;
typedef __attribute__((address_space(3))) void as3_void;
typedef __attribute__((address_space(1))) v4i  as1_v4i;
typedef __attribute__((address_space(3))) v4i  as3_v4i;

__device__ __forceinline__ void asyncCopy16(const void* g, void* l) {
  __builtin_amdgcn_global_load_async_to_lds_b128(
      (as1_v4i*)(as1_void*)g, (as3_v4i*)(as3_void*)l, 0, 0);
}
#endif

__device__ __forceinline__ u16 f2bf(float f) { // f32 -> bf16 bits, RNE
  uint32_t u = __float_as_uint(f);
  u += 0x7FFFu + ((u >> 16) & 1u);
  return (u16)(u >> 16);
}

// =======================================================================
// GEMM: C[M,N] (+epilogue) = A[M,K] * Bt[N,K]^T   (A, Bt bf16 row-major)
// block = 256 threads (8 wave32), tile 128x128, K-step 32, double-buffered
// LDS (one s_barrier per K-step). Global->LDS via async-to-LDS when the
// toolchain exposes it, else reg-staged ds_store fallback.
// wave grid 2x4 -> each wave owns 64x32 = 4x2 tiles of 16x16.
// EPI: 0 = f32 store, 1 = bf16 store, 2 = +cw[row]+qw[col]+mask, f32 store
// =======================================================================
#define BM 128
#define BN 128
#define BK 32
#define LDT 40   // LDS row stride (bf16 elems): 80B -> 16B-aligned, spread banks

template<int EPI>
__global__ __launch_bounds__(256)
void gemm_bf16(const u16* __restrict__ A, ll sA,
               const u16* __restrict__ Bt, ll sB,
               void* __restrict__ Cout, ll sC, int ldc,
               int M, int N, int K,
               const float* __restrict__ cw,
               const float* __restrict__ qw,
               const unsigned char* __restrict__ msk)
{
  __shared__ alignas(16) u16 lsA[2][BM * LDT];
  __shared__ alignas(16) u16 lsB[2][BN * LDT];

  const int tid   = threadIdx.x;
  const int lane  = tid & 31;
  const int wave  = tid >> 5;
  const int wm    = wave >> 2;          // 0..1
  const int wn    = wave & 3;           // 0..3
  const int z     = blockIdx.z;
  const int tileM = blockIdx.y * BM;
  const int tileN = blockIdx.x * BN;

  const u16* Ab = A  + (size_t)z * sA;
  const u16* Bb = Bt + (size_t)z * sB;

  // per-thread cooperative-load coordinates (2 x 16B chunks per array)
  const int row0 = tid >> 2;            // chunks 0..255
  const int seg0 = tid & 3;
  const int row1 = (tid + 256) >> 2;    // chunks 256..511
  const int seg1 = tid & 3;

  auto loadTiles = [&](int k0, int buf) {
    const B16* pa0 = reinterpret_cast<const B16*>(Ab + (size_t)(tileM + row0) * K + k0) + seg0;
    const B16* pb0 = reinterpret_cast<const B16*>(Bb + (size_t)(tileN + row0) * K + k0) + seg0;
    const B16* pa1 = reinterpret_cast<const B16*>(Ab + (size_t)(tileM + row1) * K + k0) + seg1;
    const B16* pb1 = reinterpret_cast<const B16*>(Bb + (size_t)(tileN + row1) * K + k0) + seg1;
    u16* la0 = &lsA[buf][row0 * LDT + seg0 * 8];
    u16* lb0 = &lsB[buf][row0 * LDT + seg0 * 8];
    u16* la1 = &lsA[buf][row1 * LDT + seg1 * 8];
    u16* lb1 = &lsB[buf][row1 * LDT + seg1 * 8];
#if USE_ASYNC_LDS
    asyncCopy16(pa0, la0);
    asyncCopy16(pb0, lb0);
    asyncCopy16(pa1, la1);
    asyncCopy16(pb1, lb1);
#else
    *reinterpret_cast<B16*>(la0) = *pa0;
    *reinterpret_cast<B16*>(lb0) = *pb0;
    *reinterpret_cast<B16*>(la1) = *pa1;
    *reinterpret_cast<B16*>(lb1) = *pb1;
#endif
  };

  v8f acc[4][2];
  #pragma unroll
  for (int i = 0; i < 4; ++i)
    #pragma unroll
    for (int j = 0; j < 2; ++j)
      #pragma unroll
      for (int v = 0; v < 8; ++v) acc[i][j][v] = 0.0f;

  const int mBase = wm * 64;
  const int nBase = wn * 32;
  const int r16   = lane & 15;
  // 16-bit A frag (16x32): lanes<16 hold K[0..7]+K[16..23], lanes>=16 K[8..15]+K[24..31]
  const int koffA = (lane < 16) ? 0 : 8;
  // 16-bit B frag (32x16): lanes<16 hold K[0..7]+K[16..23], lanes>=16 K[8..15]+K[24..31]
  const int koffB = (lane < 16) ? 0 : 8;

  loadTiles(0, 0);                       // prologue: tile 0 -> buffer 0

  const int nk = K / BK;
  for (int kk = 0; kk < nk; ++kk) {
    const int buf = kk & 1;
#if USE_ASYNC_LDS
    WAIT_ASYNC();                        // our async writes into buf landed
#endif
    __syncthreads();                     // everyone's writes landed / reads of buf^1 done

    if (kk + 1 < nk) loadTiles((kk + 1) * BK, buf ^ 1);

    Frag16 afr[4], bfr[2];
    #pragma unroll
    for (int mi = 0; mi < 4; ++mi) {
      const u16* p = &lsA[buf][(size_t)(mBase + mi * 16 + r16) * LDT + koffA];
      afr[mi].b[0] = *reinterpret_cast<const B16*>(p);
      afr[mi].b[1] = *reinterpret_cast<const B16*>(p + 16);
    }
    #pragma unroll
    for (int ni = 0; ni < 2; ++ni) {
      const u16* p = &lsB[buf][(size_t)(nBase + ni * 16 + r16) * LDT + koffB];
      bfr[ni].b[0] = *reinterpret_cast<const B16*>(p);
      bfr[ni].b[1] = *reinterpret_cast<const B16*>(p + 16);
    }
    #pragma unroll
    for (int mi = 0; mi < 4; ++mi)
      #pragma unroll
      for (int ni = 0; ni < 2; ++ni)
        acc[mi][ni] = __builtin_amdgcn_wmma_f32_16x16x32_bf16(
            false, afr[mi].v, false, bfr[ni].v, (short)0, acc[mi][ni], false, false);
  }

  // D layout (16x16 f32): VGPR v -> M = v (lanes 0-15) / v+8 (lanes 16-31), N = lane%16
  float* Cf = reinterpret_cast<float*>(Cout);
  u16*   Cb = reinterpret_cast<u16*>(Cout);
  const int nc   = lane & 15;
  const int rsel = (lane >> 4) << 3;
  const float* cwz = (EPI == 2) ? (cw + (size_t)z * M) : nullptr;
  const float* qwz = (EPI == 2) ? (qw + (size_t)z * N) : nullptr;
  const unsigned char* mz = (EPI == 2) ? (msk + (size_t)z * (size_t)M * N) : nullptr;

  #pragma unroll
  for (int mi = 0; mi < 4; ++mi) {
    #pragma unroll
    for (int ni = 0; ni < 2; ++ni) {
      int gn  = tileN + nBase + ni * 16 + nc;
      int gm0 = tileM + mBase + mi * 16 + rsel;
      #pragma unroll
      for (int v = 0; v < 8; ++v) {
        int row = gm0 + v;
        float val = acc[mi][ni][v];
        size_t cidx = (size_t)z * sC + (size_t)row * ldc + gn;
        if constexpr (EPI == 0) {
          Cf[cidx] = val;
        } else if constexpr (EPI == 1) {
          Cb[cidx] = f2bf(val);
        } else {
          val += cwz[row] + qwz[gn];
          val += mz[(size_t)row * N + gn] ? 0.0f : -1.0e9f;
          Cf[cidx] = val;
        }
      }
    }
  }
}

// ======================= helper kernels =======================

// f32 -> bf16 elementwise
__global__ void convert_bf16(const float* __restrict__ src, u16* __restrict__ dst, size_t n) {
  size_t i = (size_t)blockIdx.x * blockDim.x + threadIdx.x;
  if (i < n) dst[i] = f2bf(src[i]);
}

// per-row: bf16 copy + dot with weight vector (1 wave = 1 row of D)
__global__ void rowprep(const float* __restrict__ src, const float* __restrict__ w,
                        u16* __restrict__ dstB, float* __restrict__ dot, int D) {
  int wave = threadIdx.x >> 5, lane = threadIdx.x & 31;
  size_t row = (size_t)blockIdx.x * 8 + wave;
  float acc = 0.0f;
  for (int j = lane; j < D; j += 32) {
    float v = src[row * D + j];
    dstB[row * D + j] = f2bf(v);
    acc += v * w[j];
  }
  #pragma unroll
  for (int off = 16; off > 0; off >>= 1) acc += __shfl_xor(acc, off, 32);
  if (lane == 0) dot[row] = acc;
}

// batched transpose f32[R,Cc] -> bf16[Cc,R], 32x32 LDS tiles
__global__ void transpose_b(const float* __restrict__ in, u16* __restrict__ out, int R, int Cc) {
  __shared__ float tl[32][33];
  int b = blockIdx.z;
  const float* iz = in  + (size_t)b * R * Cc;
  u16*         oz = out + (size_t)b * R * Cc;
  int c0 = blockIdx.x * 32, r0 = blockIdx.y * 32;
  int tx = threadIdx.x & 31, ty = threadIdx.x >> 5;
  #pragma unroll
  for (int i = 0; i < 4; ++i)
    tl[ty + i * 8][tx] = iz[(size_t)(r0 + ty + i * 8) * Cc + c0 + tx];
  __syncthreads();
  #pragma unroll
  for (int i = 0; i < 4; ++i)
    oz[(size_t)(c0 + ty + i * 8) * R + r0 + tx] = f2bf(tl[tx][ty + i * 8]);
}

// row softmax over Q=512 (one 256-thread block per row), bf16 out
__global__ void rowsoftmax(const float* __restrict__ S, u16* __restrict__ P, int Q) {
  __shared__ float red[9];
  size_t row = blockIdx.x;
  int tid = threadIdx.x;
  const float* r = S + row * Q;
  float x0 = r[tid], x1 = r[tid + 256];
  float m = fmaxf(x0, x1);
  #pragma unroll
  for (int off = 16; off > 0; off >>= 1) m = fmaxf(m, __shfl_xor(m, off, 32));
  if ((tid & 31) == 0) red[tid >> 5] = m;
  __syncthreads();
  if (tid == 0) { float mm = red[0]; for (int i = 1; i < 8; ++i) mm = fmaxf(mm, red[i]); red[8] = mm; }
  __syncthreads();
  m = red[8];
  float e0 = __expf(x0 - m), e1 = __expf(x1 - m);
  float s = e0 + e1;
  #pragma unroll
  for (int off = 16; off > 0; off >>= 1) s += __shfl_xor(s, off, 32);
  __syncthreads();
  if ((tid & 31) == 0) red[tid >> 5] = s;
  __syncthreads();
  if (tid == 0) { float ss = 0.f; for (int i = 0; i < 8; ++i) ss += red[i]; red[8] = ss; }
  __syncthreads();
  float inv = 1.0f / red[8];
  P[row * Q + tid]       = f2bf(e0 * inv);
  P[row * Q + tid + 256] = f2bf(e1 * inv);
}

// column softmax pass 1: online max/sum over a C-chunk, coalesced along q
__global__ void colstats(const float* __restrict__ S, float* __restrict__ mp,
                         float* __restrict__ sp, int C, int Q, int chunk) {
  int q  = blockIdx.x * blockDim.x + threadIdx.x;
  int b  = blockIdx.y;
  int ch = blockIdx.z;
  const float* Sz = S + (size_t)b * C * Q;
  float m = -3.0e38f, s = 0.0f;
  int c0 = ch * chunk;
  for (int c = c0; c < c0 + chunk; ++c) {
    float x  = Sz[(size_t)c * Q + q];
    float nm = fmaxf(m, x);
    s = s * __expf(m - nm) + __expf(x - nm);
    m = nm;
  }
  size_t o = ((size_t)ch * gridDim.y + b) * Q + q;
  mp[o] = m; sp[o] = s;
}

// column softmax pass 2: combine chunk stats
__global__ void colfinal(const float* __restrict__ mp, const float* __restrict__ sp,
                         float* __restrict__ colM, float* __restrict__ colR, int BQ, int NC) {
  int i = blockIdx.x * blockDim.x + threadIdx.x;
  if (i >= BQ) return;
  float m = -3.0e38f;
  for (int c = 0; c < NC; ++c) m = fmaxf(m, mp[(size_t)c * BQ + i]);
  float s = 0.0f;
  for (int c = 0; c < NC; ++c) s += sp[(size_t)c * BQ + i] * __expf(mp[(size_t)c * BQ + i] - m);
  colM[i] = m; colR[i] = 1.0f / s;
}

// column softmax pass 3: normalize + transpose -> q2c^T [B,Q,C] bf16
__global__ void transnorm(const float* __restrict__ S, const float* __restrict__ colM,
                          const float* __restrict__ colR, u16* __restrict__ q2cT,
                          int C, int Q) {
  __shared__ float tl[32][33];
  int b = blockIdx.z;
  int c0 = blockIdx.y * 32, q0 = blockIdx.x * 32;
  int tx = threadIdx.x & 31, ty = threadIdx.x >> 5;
  const float* Sz = S + (size_t)b * C * Q;
  float cm = colM[(size_t)b * Q + q0 + tx];
  float cr = colR[(size_t)b * Q + q0 + tx];
  #pragma unroll
  for (int i = 0; i < 4; ++i) {
    float x = Sz[(size_t)(c0 + ty + i * 8) * Q + q0 + tx];
    tl[ty + i * 8][tx] = __expf(x - cm) * cr;
  }
  __syncthreads();
  u16* Oz = q2cT + (size_t)b * Q * C;
  #pragma unroll
  for (int i = 0; i < 4; ++i)
    Oz[(size_t)(q0 + ty + i * 8) * C + c0 + tx] = f2bf(tl[tx][ty + i * 8]);
}

// final: fill slot0 = context, slot2 = context*ctq, slot3 = context*qtc (in place)
__global__ void concat_out(const float* __restrict__ ctx, float* __restrict__ out,
                           int D, size_t total) {
  size_t i = (size_t)blockIdx.x * blockDim.x + threadIdx.x;
  if (i >= total) return;
  int d = (int)(i % D);
  size_t bc = i / D;
  float cv = ctx[i];
  size_t ob = bc * (size_t)(4 * D);
  out[ob + d] = cv;
  float t1 = out[ob + D + d];        // ctq (stays as slot1)
  out[ob + 2 * D + d] = cv * t1;
  float t2 = out[ob + 3 * D + d];    // qtc (overwritten with product)
  out[ob + 3 * D + d] = cv * t2;
}

// =======================================================================
extern "C" void kernel_launch(void* const* d_in, const int* in_sizes, int n_in,
                              void* d_out, int out_size, void* d_ws, size_t ws_size,
                              hipStream_t stream) {
  (void)in_sizes; (void)n_in; (void)out_size; (void)ws_size;
  const int B = 8, Q = 512, C = 2048, D = 512;

  const float* qry = (const float*)d_in[0];          // [B,Q,D]
  const float* ctx = (const float*)d_in[1];          // [B,C,D]
  const float* qwv = (const float*)d_in[2];          // [D]
  const float* cwv = (const float*)d_in[3];          // [D]
  const float* W   = (const float*)d_in[4];          // [D,D]
  const unsigned char* mask = (const unsigned char*)d_in[5]; // [B,C,Q] bool
  float* out = (float*)d_out;                        // [B,C,4D]

  // ---- carve workspace ----
  char* wsb = (char*)d_ws;
  size_t off = 0;
  auto take = [&](size_t bytes) { char* p = wsb + off; off += (bytes + 255) & ~(size_t)255; return p; };
  u16*   wb    = (u16*)  take((size_t)D * D * 2);
  u16*   cb    = (u16*)  take((size_t)B * C * D * 2);
  u16*   qb    = (u16*)  take((size_t)B * Q * D * 2);
  u16*   cbT   = (u16*)  take((size_t)B * C * D * 2);
  u16*   qbT   = (u16*)  take((size_t)B * Q * D * 2);
  u16*   tq    = (u16*)  take((size_t)B * Q * D * 2);
  float* cwd   = (float*)take((size_t)B * C * 4);
  float* qwd   = (float*)take((size_t)B * Q * 4);
  float* Sb    = (float*)take((size_t)B * C * Q * 4);
  u16*   c2q   = (u16*)  take((size_t)B * C * Q * 2);
  u16*   q2cT  = (u16*)  take((size_t)B * Q * C * 2);
  float* mpart = (float*)take((size_t)8 * B * Q * 4);
  float* spart = (float*)take((size_t)8 * B * Q * 4);
  float* colM  = (float*)take((size_t)B * Q * 4);
  float* colR  = (float*)take((size_t)B * Q * 4);
  u16*   M1T   = (u16*)  take((size_t)B * D * Q * 2);

  // ---- stage 1: conversions / row dots / transposes ----
  convert_bf16<<<(D * D) / 256, 256, 0, stream>>>(W, wb, (size_t)D * D);
  rowprep<<<(B * C) / 8, 256, 0, stream>>>(ctx, cwv, cb, cwd, D);
  rowprep<<<(B * Q) / 8, 256, 0, stream>>>(qry, qwv, qb, qwd, D);
  transpose_b<<<dim3(D / 32, C / 32, B), 256, 0, stream>>>(ctx, cbT, C, D);
  transpose_b<<<dim3(D / 32, Q / 32, B), 256, 0, stream>>>(qry, qbT, Q, D);

  // ---- stage 2: tq[q,d] = sum_e query[q,e]*W[d,e]  (A=qb, Bt=W) ----
  gemm_bf16<1><<<dim3(D / 128, Q / 128, B), 256, 0, stream>>>(
      qb, (ll)Q * D, wb, 0LL, tq, (ll)Q * D, D, Q, D, D, nullptr, nullptr, nullptr);

  // ---- stage 3: S[c,q] = context·tq + cw[c] + qw[q] + mask ----
  gemm_bf16<2><<<dim3(Q / 128, C / 128, B), 256, 0, stream>>>(
      cb, (ll)C * D, tq, (ll)Q * D, Sb, (ll)C * Q, Q, C, Q, D, cwd, qwd, mask);

  // ---- stage 4: dual softmax ----
  rowsoftmax<<<B * C, 256, 0, stream>>>(Sb, c2q, Q);
  colstats<<<dim3(Q / 256, B, 8), 256, 0, stream>>>(Sb, mpart, spart, C, Q, C / 8);
  colfinal<<<(B * Q) / 256, 256, 0, stream>>>(mpart, spart, colM, colR, B * Q, 8);
  transnorm<<<dim3(Q / 32, C / 32, B), 256, 0, stream>>>(Sb, colM, colR, q2cT, C, Q);

  // ---- stage 5: M1T[d,q] = sum_c context[c,d]*q2c[c,q]  (A=cbT, Bt=q2cT) ----
  gemm_bf16<1><<<dim3(Q / 128, D / 128, B), 256, 0, stream>>>(
      cbT, (ll)D * C, q2cT, (ll)Q * C, M1T, (ll)D * Q, Q, D, Q, C, nullptr, nullptr, nullptr);

  // ---- stage 6: ctq -> out slot1; qtc -> out slot3 (strided epilogue) ----
  gemm_bf16<0><<<dim3(D / 128, C / 128, B), 256, 0, stream>>>(
      c2q, (ll)C * Q, qbT, (ll)D * Q, out + D, (ll)C * 4 * D, 4 * D, C, D, Q,
      nullptr, nullptr, nullptr);
  gemm_bf16<0><<<dim3(D / 128, C / 128, B), 256, 0, stream>>>(
      c2q, (ll)C * Q, M1T, (ll)D * Q, out + 3 * D, (ll)C * 4 * D, 4 * D, C, D, Q,
      nullptr, nullptr, nullptr);

  // ---- stage 7: assemble remaining output slots ----
  concat_out<<<(int)(((size_t)B * C * D) / 256), 256, 0, stream>>>(
      ctx, out, D, (size_t)B * C * D);
}
